// RecursiveEncoder_45707041964142
// MI455X (gfx1250) — matrix operand order repaired and checked by reading, not compiled, not run on hardware
//
#include <hip/hip_runtime.h>

typedef __attribute__((ext_vector_type(16))) _Float16 v16h;
typedef __attribute__((ext_vector_type(8)))  _Float16 v8h;
typedef __attribute__((ext_vector_type(8)))  float    v8f;

#define M_NODES 4096
#define E_EDGES 131072
#define F_DIM   256
#define H_DIM   256
#define ET_DIM  4
#define N_ITERS 2
#define K_EDGE  516          // 2H + ET
#define KT_EDGE 17           // ceil(516/32) -> K padded to 544
#define KT_CHILD 8           // 256/32
#define NT      16           // 256/16 N-tiles
#define EDGE_ROWS 32         // edges per block (2 WMMA row tiles, B-fragment reuse x2)
#define S_EDGE  552          // LDS row stride in halves; 276 dwords == 20 mod 64 -> conflict-free

// ---------------------------------------------------------------- zero init
__global__ void k_zero(unsigned int* p, int n) {
    int i = blockIdx.x * blockDim.x + threadIdx.x;
    if (i < n) p[i] = 0u;
}

// ------------------------------------------------- pack weights to B-fragment
// layout: frag[((kt*NT + nt)*32 + lane)*16 + h] = W[kt*32 + (lane<16?0:8) + (h<8?h:h+8)][nt*16 + lane%16]
__global__ void k_pack(const float* __restrict__ childW, const float* __restrict__ neW,
                       _Float16* __restrict__ cw16, _Float16* __restrict__ ne16) {
    int tid = blockIdx.x * blockDim.x + threadIdx.x;
    const int CH = KT_CHILD * NT * 512;              // 65536 halves
    const int NE = N_ITERS * KT_EDGE * NT * 512;     // 278528 halves
    if (tid < CH) {
        int h = tid & 15, lane = (tid >> 4) & 31, ntkt = tid >> 9;
        int nt = ntkt & 15, kt = ntkt >> 4;
        int col = nt * 16 + (lane & 15);
        int k   = kt * 32 + ((lane < 16) ? 0 : 8) + ((h < 8) ? h : h + 8);
        cw16[tid] = (_Float16)childW[(size_t)k * H_DIM + col];
    } else if (tid < CH + NE) {
        int id = tid - CH;
        const int PER = KT_EDGE * NT * 512;          // 139264 halves / iter
        int it = id / PER, rem = id % PER;
        int h = rem & 15, lane = (rem >> 4) & 31, ntkt = rem >> 9;
        int nt = ntkt & 15, kt = ntkt >> 4;
        int col = nt * 16 + (lane & 15);
        int k   = kt * 32 + ((lane < 16) ? 0 : 8) + ((h < 8) ? h : h + 8);
        float v = (k < K_EDGE) ? neW[(size_t)it * K_EDGE * H_DIM + (size_t)k * H_DIM + col] : 0.0f;
        ne16[id] = (_Float16)v;
    }
}

// A-fragment from an LDS tile row-major with stride S (halves).
// ISA 16-bit A 16x32 layout: lane<16 holds K {0..7,16..23}, lane>=16 holds K {8..15,24..31}.
static __device__ inline v16h load_a(const _Float16* ap) {
    v8h lo = *(const v8h*)ap;
    v8h hi = *(const v8h*)(ap + 16);
    v16h a;
#pragma unroll
    for (int i = 0; i < 8; ++i) { a[i] = lo[i]; a[i + 8] = hi[i]; }
    return a;
}

// ------------------------------------- child projection: cf = relu(X@W+b)*exists
__global__ __launch_bounds__(256) void k_child(const float* __restrict__ feats,
                                               const float* __restrict__ exists,
                                               const float* __restrict__ bias,
                                               const _Float16* __restrict__ cw16,
                                               _Float16* __restrict__ cf16,
                                               unsigned int* __restrict__ parent0) {
    __shared__ __align__(16) _Float16 xs[16 * 256];
    int t = threadIdx.x;
    int rowBase = blockIdx.x * 16;
    {   // stage 16x256 f32 rows -> f16 LDS
        int r = t >> 4, sub = t & 15;
        const float* g = feats + (size_t)(rowBase + r) * F_DIM + sub * 16;
        _Float16* d = xs + r * 256 + sub * 16;
#pragma unroll
        for (int j = 0; j < 16; ++j) d[j] = (_Float16)g[j];
    }
    __syncthreads();
    int wave = t >> 5, lane = t & 31;
    int rloc = lane & 15, k0 = (lane < 16) ? 0 : 8;
    int nt0 = wave * 2, nt1 = nt0 + 1;
    v8f c0 = {}, c1 = {};
#pragma unroll
    for (int kt = 0; kt < KT_CHILD; ++kt) {
        v16h a  = load_a(xs + rloc * 256 + kt * 32 + k0);
        v16h b0 = *(const v16h*)(cw16 + ((size_t)(kt * NT + nt0) * 32 + lane) * 16);
        v16h b1 = *(const v16h*)(cw16 + ((size_t)(kt * NT + nt1) * 32 + lane) * 16);
        c0 = __builtin_amdgcn_wmma_f32_16x16x32_f16(false, a, false, b0, (short)0, c0, false, false);
        c1 = __builtin_amdgcn_wmma_f32_16x16x32_f16(false, a, false, b1, (short)0, c1, false, false);
    }
    int mBase = (lane < 16) ? 0 : 8;
#pragma unroll
    for (int h2 = 0; h2 < 2; ++h2) {
        v8f c = h2 ? c1 : c0;
        int n = (h2 ? nt1 : nt0) * 16 + (lane & 15);
        float b = bias[n];
        float pm = 0.0f;
#pragma unroll
        for (int v = 0; v < 8; ++v) {
            int row = rowBase + mBase + v;
            float val = fmaxf(c[v] + b, 0.0f) * exists[row];
            cf16[(size_t)row * H_DIM + n] = (_Float16)val;
            pm = fmaxf(pm, val);
        }
        pm = fmaxf(pm, __shfl_xor(pm, 16, 32));          // merge M 0..7 with 8..15
        if (lane < 16) atomicMax(parent0 + n, __float_as_uint(pm));  // vals >= 0
    }
}

// ------- edge MLP + fused segment-max: ncf[from] = max(ncf[from], relu(x@W+b))
// 32 edges per block: each B fragment feeds 2 row-tile WMMAs (halved L2 weight traffic)
__global__ __launch_bounds__(256) void k_edge(const _Float16* __restrict__ cf16,
                                              const int* __restrict__ eidx,
                                              const float* __restrict__ ef,
                                              const float* __restrict__ bias,
                                              const _Float16* __restrict__ w16,
                                              unsigned int* __restrict__ ncf) {
    __shared__ __align__(16) _Float16 xs[EDGE_ROWS * S_EDGE];
    __shared__ int sfrom[EDGE_ROWS];
    int t = threadIdx.x;
    int eBase = blockIdx.x * EDGE_ROWS;
    {   // gather 32 edge rows: [cf16[from] | cf16[to] | ef | 0-pad] -> LDS f16 (b128 copies)
        int r = t >> 3, sub = t & 7;
        int e = eBase + r;
        int from = eidx[2 * e], to = eidx[2 * e + 1];
        uint4* dst = (uint4*)(xs + r * S_EDGE);
        const uint4* sf = (const uint4*)(cf16 + (size_t)from * H_DIM);
        const uint4* st = (const uint4*)(cf16 + (size_t)to * H_DIM);
#pragma unroll
        for (int j = 0; j < 4; ++j) dst[sub * 4 + j] = sf[sub * 4 + j];
#pragma unroll
        for (int j = 0; j < 4; ++j) dst[32 + sub * 4 + j] = st[sub * 4 + j];
        if (sub == 0) {
            sfrom[r] = from;
            unsigned int* dw = (unsigned int*)(xs + r * S_EDGE);
#pragma unroll
            for (int j = 258; j < 272; ++j) dw[j] = 0u;    // zero K 516..543
#pragma unroll
            for (int j = 0; j < ET_DIM; ++j)
                xs[r * S_EDGE + 512 + j] = (_Float16)ef[(size_t)e * ET_DIM + j];
        }
    }
    __syncthreads();
    int wave = t >> 5, lane = t & 31;
    int rloc = lane & 15, k0 = (lane < 16) ? 0 : 8;
    int nt0 = wave * 2, nt1 = nt0 + 1;
    v8f c00 = {}, c01 = {}, c10 = {}, c11 = {};
    for (int kt = 0; kt < KT_EDGE; ++kt) {
        v16h a0 = load_a(xs + rloc * S_EDGE + kt * 32 + k0);
        v16h a1 = load_a(xs + (16 + rloc) * S_EDGE + kt * 32 + k0);
        v16h b0 = *(const v16h*)(w16 + ((size_t)(kt * NT + nt0) * 32 + lane) * 16);
        v16h b1 = *(const v16h*)(w16 + ((size_t)(kt * NT + nt1) * 32 + lane) * 16);
        c00 = __builtin_amdgcn_wmma_f32_16x16x32_f16(false, a0, false, b0, (short)0, c00, false, false);
        c01 = __builtin_amdgcn_wmma_f32_16x16x32_f16(false, a0, false, b1, (short)0, c01, false, false);
        c10 = __builtin_amdgcn_wmma_f32_16x16x32_f16(false, a1, false, b0, (short)0, c10, false, false);
        c11 = __builtin_amdgcn_wmma_f32_16x16x32_f16(false, a1, false, b1, (short)0, c11, false, false);
    }
    int mBase = (lane < 16) ? 0 : 8;
#pragma unroll
    for (int mt = 0; mt < 2; ++mt) {
#pragma unroll
        for (int h2 = 0; h2 < 2; ++h2) {
            v8f c = mt ? (h2 ? c11 : c10) : (h2 ? c01 : c00);
            int n = (h2 ? nt1 : nt0) * 16 + (lane & 15);
            float b = bias[n];
#pragma unroll
            for (int v = 0; v < 8; ++v) {
                float val = fmaxf(c[v] + b, 0.0f);         // >= 0 -> uint max valid
                int row = sfrom[mt * 16 + mBase + v];
                atomicMax(ncf + (size_t)row * H_DIM + n, __float_as_uint(val));
            }
        }
    }
}

// ------- per-iteration post: column max -> parent slice; re-emit cf as f16
__global__ void k_post(const float* __restrict__ ncf, _Float16* __restrict__ cf16,
                       unsigned int* __restrict__ parentSlice) {
    int t = threadIdx.x;
    int rowBase = blockIdx.x * 16;
    float mx = 0.0f;
#pragma unroll
    for (int r = 0; r < 16; ++r) {
        float v = ncf[(size_t)(rowBase + r) * H_DIM + t];
        cf16[(size_t)(rowBase + r) * H_DIM + t] = (_Float16)v;
        mx = fmaxf(mx, v);
    }
    atomicMax(parentSlice + t, __float_as_uint(mx));
}

// ------- tiny head: out = relu(parent(1x768) @ W(768x256) + b)
__global__ void k_final(const unsigned int* __restrict__ parentBits,
                        const float* __restrict__ pW, const float* __restrict__ pB,
                        float* __restrict__ out) {
    __shared__ float sp[768];
    int t = threadIdx.x;
#pragma unroll
    for (int j = 0; j < 3; ++j) sp[t + j * 256] = __uint_as_float(parentBits[t + j * 256]);
    __syncthreads();
    float acc = 0.0f;
    for (int k = 0; k < 768; ++k) acc += sp[k] * pW[(size_t)k * 256 + t];
    out[t] = fmaxf(acc + pB[t], 0.0f);
}

extern "C" void kernel_launch(void* const* d_in, const int* in_sizes, int n_in,
                              void* d_out, int out_size, void* d_ws, size_t ws_size,
                              hipStream_t stream) {
    const float* child_feats = (const float*)d_in[0];
    const float* child_exist = (const float*)d_in[1];
    const float* edge_onehot = (const float*)d_in[2];
    const int*   edge_idx    = (const int*)d_in[3];
    const float* child_W     = (const float*)d_in[4];
    const float* child_b     = (const float*)d_in[5];
    const float* ne_W        = (const float*)d_in[6];
    const float* ne_b        = (const float*)d_in[7];
    const float* parent_W    = (const float*)d_in[8];
    const float* parent_b    = (const float*)d_in[9];
    float* out = (float*)d_out;

    // workspace carve-up (all offsets 256B aligned)
    char* ws = (char*)d_ws;
    _Float16* cw16 = (_Float16*)(ws);                         // 131072 B
    _Float16* ne16 = (_Float16*)(ws + 131072);                // 557056 B
    _Float16* cf16 = (_Float16*)(ws + 688128);                // 2 MB
    unsigned int* ncf0   = (unsigned int*)(ws + 2785280);     // 4 MB
    unsigned int* ncf1   = (unsigned int*)(ws + 6979584);     // 4 MB
    unsigned int* parent = (unsigned int*)(ws + 11173888);    // 3072 B

    const int NCF = M_NODES * H_DIM;                          // 1048576
    k_zero<<<(NCF + 255) / 256, 256, 0, stream>>>(ncf0, NCF);
    k_zero<<<(NCF + 255) / 256, 256, 0, stream>>>(ncf1, NCF);
    k_zero<<<(768 + 255) / 256, 256, 0, stream>>>(parent, 768);

    const int PACK = KT_CHILD * NT * 512 + N_ITERS * KT_EDGE * NT * 512;
    k_pack<<<(PACK + 255) / 256, 256, 0, stream>>>(child_W, ne_W, cw16, ne16);

    k_child<<<M_NODES / 16, 256, 0, stream>>>(child_feats, child_exist, child_b,
                                              cw16, cf16, parent);

    const size_t NE_PER = (size_t)KT_EDGE * NT * 512;         // halves per iter
    // iter 0
    k_edge<<<E_EDGES / EDGE_ROWS, 256, 0, stream>>>(cf16, edge_idx, edge_onehot,
                                                    ne_b, ne16, ncf0);
    k_post<<<M_NODES / 16, 256, 0, stream>>>((const float*)ncf0, cf16, parent + 256);
    // iter 1
    k_edge<<<E_EDGES / EDGE_ROWS, 256, 0, stream>>>(cf16, edge_idx, edge_onehot,
                                                    ne_b + 256, ne16 + NE_PER, ncf1);
    k_post<<<M_NODES / 16, 256, 0, stream>>>((const float*)ncf1, cf16, parent + 512);

    k_final<<<1, 256, 0, stream>>>(parent, parent_W, parent_b, out);
}
